// EdgeAwareGNN_9964324126965
// MI455X (gfx1250) — matrix operand order repaired and checked by reading, not compiled, not run on hardware
//
#include <hip/hip_runtime.h>
#include <hip/hip_bf16.h>

// ---------------------------------------------------------------------------
// EdgeAwareGNN on MI455X (gfx1250): bf16 WMMA for the edge-MLP GEMMs, fused
// with the per-edge matvec + scatter so per-edge weight tensors are never
// materialized. wave32, 16x16x32 bf16 WMMA tiles, A synthesized in registers.
// ---------------------------------------------------------------------------

#define STIM_SCALE_F (1.0f / (0.0066f * 2.0f))

constexpr int cIN_F  = 6;
constexpr int cHID   = 64;
constexpr int cN_SRC1 = 65536;
constexpr int cN_DST1 = 32768;
constexpr int cN_DST2 = 16384;
constexpr int cN_DST3 = 4096;
constexpr int cE1 = 131072;
constexpr int cE2 = 16384;
constexpr int cE3 = 32768;

typedef __attribute__((ext_vector_type(16))) __bf16 bf16x16;
typedef __attribute__((ext_vector_type(8)))  float  f32x8;

// ---------------------------------------------------------------------------
// Small prep kernels
// ---------------------------------------------------------------------------

// u[k] = wa[0,k]; v[k] = wa[1,k] + ba[k]   (edge MLP layer-a collapses to
// relu(s*u + v) because edge features are [s, 1])
__global__ void prep_uv(const float* __restrict__ wa, const float* __restrict__ ba,
                        float* __restrict__ u, float* __restrict__ v, int d) {
    int k = blockIdx.x * blockDim.x + threadIdx.x;
    if (k < d) {
        u[k] = wa[k];
        v[k] = wa[d + k] + ba[k];
    }
}

// wbT[n*d + k] = bf16(wb[k*d + n])  (n-major bf16 copy for WMMA B fragments)
__global__ void cvt_transpose_bf16(const float* __restrict__ wb,
                                   __bf16* __restrict__ wbT, int d) {
    long long idx = (long long)blockIdx.x * blockDim.x + threadIdx.x;
    long long tot = (long long)d * d;
    if (idx >= tot) return;
    int k = (int)(idx / d);
    int n = (int)(idx % d);
    wbT[(size_t)n * d + k] = (__bf16)wb[idx];
}

__global__ void zero_f32(float* __restrict__ p, long long n) {
    long long stride = (long long)gridDim.x * blockDim.x;
    for (long long i = (long long)blockIdx.x * blockDim.x + threadIdx.x; i < n; i += stride)
        p[i] = 0.0f;
}

__global__ void deg_count(const int* __restrict__ dst, float* __restrict__ deg, int E) {
    int e = blockIdx.x * blockDim.x + threadIdx.x;
    if (e < E) atomicAdd(&deg[dst[e]], 1.0f);
}

// ---------------------------------------------------------------------------
// Fused NNConv message kernel (WMMA).
//   One wave (32 threads) owns a tile of 16 edges. For each input feature i
//   and each 16-wide output column block ns, it runs the K-reduction
//   W_tile = A @ B with:
//     A[e,k] = relu(s_e * u[k] + v[k])        (synthesized in registers)
//     B[k,n] = wbT[n*D + k]                   (bf16, n-major, 2x b128/lane)
//   then immediately contracts W_tile with gathered x[src] rows and
//   accumulates the 16 x OUT_F message tile, finally atomically scattering
//   into agg[dst]. blockIdx.y splits the K range for occupancy; only the
//   kc==0 split adds the bb bias term.
// ---------------------------------------------------------------------------
template <int IN_F_, int OUT_F_>
__global__ __launch_bounds__(32)
void nnconv_msg_wmma(const float* __restrict__ x,      // [n_src, IN_F_]
                     const float* __restrict__ stim,   // [E]
                     const int*   __restrict__ src,    // [E]
                     const int*   __restrict__ dst,    // [E]
                     const float* __restrict__ u,      // [D]
                     const float* __restrict__ v,      // [D]
                     const __bf16* __restrict__ wbT,   // [D][D] n-major bf16
                     const float* __restrict__ bb,     // [D]
                     float* __restrict__ agg)          // [n_dst, OUT_F_]
{
    constexpr int D  = IN_F_ * OUT_F_;
    constexpr int NS = OUT_F_ / 16;                    // 16-col sub-tiles

    __shared__ float s_s[16];                          // scaled stimulus
    __shared__ int   s_dst[16];
    __shared__ float s_xg[16][IN_F_];                  // gathered source rows

    const int ebase = blockIdx.x * 16;
    const int lane  = threadIdx.x;                     // 0..31
    const int half  = lane >> 4;                       // 0/1
    const int col   = lane & 15;

    if (lane < 16) {
        s_s[lane]   = stim[ebase + lane] * STIM_SCALE_F;
        s_dst[lane] = dst[ebase + lane];
    }
    for (int idx = lane; idx < 16 * IN_F_; idx += 32) {
        int e = idx / IN_F_;
        int i = idx % IN_F_;
        s_xg[e][i] = x[(size_t)src[ebase + e] * IN_F_ + i];
    }
    __syncthreads();

    const int kc   = blockIdx.y;
    const int kper = D / (int)gridDim.y;               // multiple of 32
    const int kbeg = kc * kper;
    const int kend = kbeg + kper;

    // A-fragment row for this lane (A layout: lane L holds row M = L % 16)
    const float srow = s_s[col];

    const f32x8 zf = {0.f, 0.f, 0.f, 0.f, 0.f, 0.f, 0.f, 0.f};
    f32x8 msg[NS];
#pragma unroll
    for (int ns = 0; ns < NS; ++ns) msg[ns] = zf;

    for (int i = 0; i < IN_F_; ++i) {
        f32x8 wacc[NS];
#pragma unroll
        for (int ns = 0; ns < NS; ++ns) wacc[ns] = zf;

        for (int k0 = kbeg; k0 < kend; k0 += 32) {
            // Synthesize A fragment: element t of lane (half,col) holds
            // K = k0 + half*16 + t, row M = col.
            bf16x16 af;
#pragma unroll
            for (int t = 0; t < 16; ++t) {
                int k  = k0 + (half << 4) + t;
                float a = fmaf(srow, u[k], v[k]);
                a = a > 0.0f ? a : 0.0f;
                af[t] = (__bf16)a;
            }
#pragma unroll
            for (int ns = 0; ns < NS; ++ns) {
                // B fragment: element t = B[k0 + half*16 + t][n], n = col of
                // this 16-col block -> 32 contiguous bf16 from n-major wbT.
                const int n = i * OUT_F_ + ns * 16 + col;
                const bf16x16* bp =
                    (const bf16x16*)(wbT + (size_t)n * D + k0 + (half << 4));
                bf16x16 bf = *bp;
                wacc[ns] = __builtin_amdgcn_wmma_f32_16x16x32_bf16(
                    false, af, false, bf, (short)0, wacc[ns], false, false);
            }
        }

        // Epilogue: msg[e, o] += x[src_e, i] * (W[e, i*OUT_F+o] + bb)
        // C layout: element r of lane (half,col) is row M = half*8 + r, N=col.
        float xv[8];
#pragma unroll
        for (int r = 0; r < 8; ++r) xv[r] = s_xg[half * 8 + r][i];
#pragma unroll
        for (int ns = 0; ns < NS; ++ns) {
            float bbv = (kc == 0) ? bb[i * OUT_F_ + ns * 16 + col] : 0.0f;
#pragma unroll
            for (int r = 0; r < 8; ++r)
                msg[ns][r] += xv[r] * (wacc[ns][r] + bbv);
        }
    }

    // Scatter-add message tile into destination aggregation buffer.
#pragma unroll
    for (int ns = 0; ns < NS; ++ns) {
#pragma unroll
        for (int r = 0; r < 8; ++r) {
            int e = half * 8 + r;
            atomicAdd(&agg[(size_t)s_dst[e] * OUT_F_ + ns * 16 + col], msg[ns][r]);
        }
    }
}

// ---------------------------------------------------------------------------
// Mean + bias + LayerNorm + ReLU (in place). One 64-thread block per node row.
// ---------------------------------------------------------------------------
__global__ void finalize_ln(float* __restrict__ agg, const float* __restrict__ deg,
                            const float* __restrict__ bias,
                            const float* __restrict__ gamma,
                            const float* __restrict__ beta) {
    __shared__ float red[64];
    const int node = blockIdx.x;
    const int t    = threadIdx.x;   // 0..63
    float d = deg[node];
    d = d > 1.0f ? d : 1.0f;
    float h = agg[(size_t)node * 64 + t] / d + bias[t];

    red[t] = h;
    __syncthreads();
    for (int off = 32; off > 0; off >>= 1) {
        if (t < off) red[t] += red[t + off];
        __syncthreads();
    }
    float mu = red[0] * (1.0f / 64.0f);
    __syncthreads();
    float c = h - mu;
    red[t] = c * c;
    __syncthreads();
    for (int off = 32; off > 0; off >>= 1) {
        if (t < off) red[t] += red[t + off];
        __syncthreads();
    }
    float var = red[0] * (1.0f / 64.0f);
    float y = c * rsqrtf(var + 1e-5f) * gamma[t] + beta[t];
    agg[(size_t)node * 64 + t] = y > 0.0f ? y : 0.0f;
}

// ---------------------------------------------------------------------------
// Layer 3 (out_f = 1): per-edge W column (64) + dot with gathered row.
// One wave per edge; shuffle reduction; atomic scatter of the scalar.
// ---------------------------------------------------------------------------
__global__ void nnconv3_msg(const float* __restrict__ h2,   // [N_DST2, 64]
                            const float* __restrict__ stim,
                            const int* __restrict__ src, const int* __restrict__ dst,
                            const float* __restrict__ u, const float* __restrict__ v,
                            const float* __restrict__ wb,   // [64,64] f32
                            const float* __restrict__ bb,   // [64]
                            float* __restrict__ agg)        // [N_DST3]
{
    const int e    = blockIdx.x * (blockDim.x >> 5) + (threadIdx.x >> 5);
    const int lane = threadIdx.x & 31;
    const float s  = stim[e] * STIM_SCALE_F;

    float acc0 = 0.0f, acc1 = 0.0f;
    for (int k = 0; k < 64; ++k) {
        float a = fmaf(s, u[k], v[k]);
        a = a > 0.0f ? a : 0.0f;
        acc0 = fmaf(a, wb[k * 64 + lane], acc0);
        acc1 = fmaf(a, wb[k * 64 + lane + 32], acc1);
    }
    acc0 += bb[lane];
    acc1 += bb[lane + 32];

    const float* xr = h2 + (size_t)src[e] * 64;
    float t = acc0 * xr[lane] + acc1 * xr[lane + 32];
    for (int off = 16; off > 0; off >>= 1) t += __shfl_down(t, off, 32);
    if (lane == 0) atomicAdd(&agg[dst[e]], t);
}

__global__ void out_softplus(const float* __restrict__ agg, const float* __restrict__ deg,
                             const float* __restrict__ bias3, float* __restrict__ out, int n) {
    int i = blockIdx.x * blockDim.x + threadIdx.x;
    if (i >= n) return;
    float d = deg[i];
    d = d > 1.0f ? d : 1.0f;
    float h = agg[i] / d + bias3[0];
    out[i] = (h > 20.0f) ? h : log1pf(expf(h));
}

// ---------------------------------------------------------------------------
// Launch
// ---------------------------------------------------------------------------
extern "C" void kernel_launch(void* const* d_in, const int* in_sizes, int n_in,
                              void* d_out, int out_size, void* d_ws, size_t ws_size,
                              hipStream_t stream) {
    const float* x     = (const float*)d_in[0];
    const float* stim1 = (const float*)d_in[1];
    const float* stim2 = (const float*)d_in[2];
    const float* stim3 = (const float*)d_in[3];
    const int*   src1  = (const int*)d_in[4];
    const int*   dst1  = (const int*)d_in[5];
    const int*   src2  = (const int*)d_in[6];
    const int*   dst2  = (const int*)d_in[7];
    const int*   src3  = (const int*)d_in[8];
    const int*   dst3  = (const int*)d_in[9];
    const float* w1a = (const float*)d_in[10];
    const float* b1a = (const float*)d_in[11];
    const float* w1b = (const float*)d_in[12];
    const float* b1b = (const float*)d_in[13];
    const float* w2a = (const float*)d_in[14];
    const float* b2a = (const float*)d_in[15];
    const float* w2b = (const float*)d_in[16];
    const float* b2b = (const float*)d_in[17];
    const float* w3a = (const float*)d_in[18];
    const float* b3a = (const float*)d_in[19];
    const float* w3b = (const float*)d_in[20];
    const float* b3b = (const float*)d_in[21];
    const float* bias1 = (const float*)d_in[22];
    const float* bias2 = (const float*)d_in[23];
    const float* bias3 = (const float*)d_in[24];
    const float* ln1_g = (const float*)d_in[25];
    const float* ln1_b = (const float*)d_in[26];
    const float* ln2_g = (const float*)d_in[27];
    const float* ln2_b = (const float*)d_in[28];
    float* out = (float*)d_out;

    // Workspace carving (256B aligned)
    char*  ws  = (char*)d_ws;
    size_t off = 0;
    auto carve = [&](size_t bytes) -> char* {
        char* p = ws + off;
        off = (off + bytes + 255) & ~(size_t)255;
        return p;
    };
    constexpr int D1 = cIN_F * cHID;   // 384
    constexpr int D2 = cHID * cHID;    // 4096
    constexpr int D3 = cHID;           // 64

    __bf16* wbT2 = (__bf16*)carve((size_t)D2 * D2 * sizeof(__bf16)); // 33.5 MB
    __bf16* wbT1 = (__bf16*)carve((size_t)D1 * D1 * sizeof(__bf16));
    float* u1 = (float*)carve(D1 * 4);
    float* v1 = (float*)carve(D1 * 4);
    float* u2 = (float*)carve(D2 * 4);
    float* v2 = (float*)carve(D2 * 4);
    float* u3 = (float*)carve(D3 * 4);
    float* v3 = (float*)carve(D3 * 4);

    size_t zstart = off;   // everything below is atomically accumulated -> zero it
    float* agg1 = (float*)carve((size_t)cN_DST1 * cHID * 4);  // also h1 (in place)
    float* agg2 = (float*)carve((size_t)cN_DST2 * cHID * 4);  // also h2 (in place)
    float* agg3 = (float*)carve((size_t)cN_DST3 * 4);
    float* deg1 = (float*)carve((size_t)cN_DST1 * 4);
    float* deg2 = (float*)carve((size_t)cN_DST2 * 4);
    float* deg3 = (float*)carve((size_t)cN_DST3 * 4);
    size_t zbytes = off - zstart;
    (void)ws_size; (void)in_sizes; (void)n_in; (void)out_size;

    // --- prep: collapse edge-MLP layer a, transpose+bf16 layer b weights ---
    prep_uv<<<(D1 + 255) / 256, 256, 0, stream>>>(w1a, b1a, u1, v1, D1);
    prep_uv<<<(D2 + 255) / 256, 256, 0, stream>>>(w2a, b2a, u2, v2, D2);
    prep_uv<<<1, 64, 0, stream>>>(w3a, b3a, u3, v3, D3);
    cvt_transpose_bf16<<<((long long)D1 * D1 + 255) / 256, 256, 0, stream>>>(w1b, wbT1, D1);
    cvt_transpose_bf16<<<((long long)D2 * D2 + 255) / 256, 256, 0, stream>>>(w2b, wbT2, D2);

    zero_f32<<<2048, 256, 0, stream>>>((float*)(ws + zstart), (long long)(zbytes / 4));

    deg_count<<<cE1 / 256, 256, 0, stream>>>(dst1, deg1, cE1);
    deg_count<<<cE2 / 256, 256, 0, stream>>>(dst2, deg2, cE2);
    deg_count<<<cE3 / 256, 256, 0, stream>>>(dst3, deg3, cE3);

    // --- layer 1: E1=131072 edges, IN_F=6, OUT_F=64, D=384, K split x4 ---
    nnconv_msg_wmma<cIN_F, cHID><<<dim3(cE1 / 16, 4), 32, 0, stream>>>(
        x, stim1, src1, dst1, u1, v1, wbT1, b1b, agg1);
    finalize_ln<<<cN_DST1, 64, 0, stream>>>(agg1, deg1, bias1, ln1_g, ln1_b);

    // --- layer 2: E2=16384 edges, IN_F=64, OUT_F=64, D=4096, K split x8 ---
    nnconv_msg_wmma<cHID, cHID><<<dim3(cE2 / 16, 8), 32, 0, stream>>>(
        agg1, stim2, src2, dst2, u2, v2, wbT2, b2b, agg2);
    finalize_ln<<<cN_DST2, 64, 0, stream>>>(agg2, deg2, bias2, ln2_g, ln2_b);

    // --- layer 3: E3=32768 edges, IN_F=64, OUT_F=1, D=64 (VALU, tiny) ---
    nnconv3_msg<<<cE3 / 8, 256, 0, stream>>>(agg2, stim3, src3, dst3,
                                             u3, v3, w3b, b3b, agg3);

    // --- output: mean + bias + softplus ---
    out_softplus<<<(cN_DST3 + 255) / 256, 256, 0, stream>>>(agg3, deg3, bias3, out, cN_DST3);
}
